// CondTransformerBlock_36601711296577
// MI455X (gfx1250) — matrix operand order repaired and checked by reading, not compiled, not run on hardware
//
#include <hip/hip_runtime.h>
#include <hip/hip_bf16.h>

// ---------------------------------------------------------------------------
// CondTransformerBlock on MI455X (gfx1250, wave32, WMMA + async/TDM staging)
//
//   K0 init        : zero agg/sum, set segment-max keys to enc(-inf)
//   K1 qkvs_gemm   : q,k,v,skip = x@W + b   (WMMA f16; x tile staged via
//                    global_load_async_to_lds_b128, W pre-swizzled f16 in LDS)
//   K2 edge_gemm   : e = edge_attr@We -> fp16 (WMMA; edge tile staged via
//                    tensor_load_to_lds (TDM) + s_wait_tensorcnt)
//   K3 logits+max  : logits = <q[tgt], k[src]+e>/4 ; atomicMax (ordered-uint)
//   K4 exp+sum     : ex = exp(l - m[tgt]) ; atomicAdd sum
//   K5 aggregate   : agg[tgt] += alpha * (v[src]+e)
//   K6 finalize    : beta gate + conditional LayerNorm + ReLU (wave/node)
//
// HBM-bound (~2 GB @ 23.3 TB/s); all GEMMs on the WMMA pipe, biggest
// intermediate (e) kept fp16.  Workspace ~362 MB.
// ---------------------------------------------------------------------------

typedef __attribute__((ext_vector_type(16))) _Float16 v16h;
typedef __attribute__((ext_vector_type(8)))  float    v8f;
typedef __attribute__((ext_vector_type(4)))  unsigned v4u;
typedef __attribute__((ext_vector_type(8)))  unsigned v8u;

#define N_NODES 50000
#define N_EDGES 800000
#define DIM     128      // F = H*D
#define HEADS   8
#define HDIM    16
#define DEDGE   32
#define DCOND   16

// ----- C/D 16x16 f32 stores (wave32 layout, ISA 7.12.2) ---------------------
__device__ __forceinline__ void store_c_f32(float* __restrict__ C, int ldc, v8f c) {
  const int lane = threadIdx.x & 31;
  const int col  = lane & 15;
  const int half = lane >> 4;
#pragma unroll
  for (int r = 0; r < 8; ++r) C[(size_t)((half << 3) + r) * ldc + col] = c[r];
}
__device__ __forceinline__ void store_c_f16(_Float16* __restrict__ C, int ldc, v8f c) {
  const int lane = threadIdx.x & 31;
  const int col  = lane & 15;
  const int half = lane >> 4;
#pragma unroll
  for (int r = 0; r < 8; ++r) C[(size_t)((half << 3) + r) * ldc + col] = (_Float16)c[r];
}

// A fragment (16x32 f16) from an fp32 row-major LDS tile, stride lda floats.
__device__ __forceinline__ v16h load_a_lds(const float* rp_row0, int lda) {
  const int lane = threadIdx.x & 31;
  const int half = lane >> 4;
  const float* rp = rp_row0 + (size_t)(lane & 15) * lda;
  v16h a;
#pragma unroll
  for (int i = 0; i < 16; ++i)
    a[i] = (_Float16)rp[(half << 4) + i];
  return a;
}

// ----- ordered-uint encoding for float atomicMax ----------------------------
__device__ __forceinline__ unsigned enc_f32(float f) {
  unsigned u = __float_as_uint(f);
  return u ^ ((u >> 31) ? 0xFFFFFFFFu : 0x80000000u);
}
__device__ __forceinline__ float dec_f32(unsigned u) {
  unsigned b = (u & 0x80000000u) ? (u ^ 0x80000000u) : ~u;
  return __uint_as_float(b);
}
#define NEG_INF_KEY 0x007FFFFFu  // enc(-inf)

__device__ __forceinline__ float wave_sum(float v) {
#pragma unroll
  for (int off = 16; off > 0; off >>= 1) v += __shfl_xor(v, off, 32);
  return v;
}

// ---------------------------------------------------------------------------
// K0: init accumulators (re-run every call; no cross-call state)
// ---------------------------------------------------------------------------
__global__ __launch_bounds__(256) void k0_init(float* __restrict__ agg,
                                               float* __restrict__ ssum,
                                               unsigned* __restrict__ mkey) {
  int idx = blockIdx.x * 256 + threadIdx.x;
  if (idx < N_NODES * DIM) agg[idx] = 0.0f;
  if (idx < N_NODES * HEADS) { ssum[idx] = 0.0f; mkey[idx] = NEG_INF_KEY; }
}

// ---------------------------------------------------------------------------
// K1: q/k/v/skip GEMM.  grid=(N/16, 4 matrices), block=256 (8 waves = 8
// column tiles).  x tile (16x128 f32, 8KB) staged into LDS with async
// load-to-LDS; W (128x128) converted once per block to f16 in LDS, laid out
// in exact WMMA B-fragment order so each lane reads 32 contiguous bytes.
// ---------------------------------------------------------------------------
__global__ __launch_bounds__(256) void k1_qkvs_gemm(
    const float* __restrict__ x,
    const float* __restrict__ Wq, const float* __restrict__ bq,
    const float* __restrict__ Wk, const float* __restrict__ bk,
    const float* __restrict__ Wv, const float* __restrict__ bv,
    const float* __restrict__ Wsk, const float* __restrict__ bsk,
    float* __restrict__ oq, float* __restrict__ ok,
    float* __restrict__ ov, float* __restrict__ osk) {
  __shared__ __align__(16) float    xs[16 * DIM];            // 8 KB
  __shared__ __align__(32) _Float16 wls[4 * 8 * 32 * 16];    // 32 KB, frag order

  const int wave = threadIdx.x >> 5;
  const int lane = threadIdx.x & 31;
  const int row0 = blockIdx.x << 4;

  const float* W; const float* bias; float* out;
  switch (blockIdx.y) {
    case 0:  W = Wq;  bias = bq;  out = oq;  break;
    case 1:  W = Wk;  bias = bk;  out = ok;  break;
    case 2:  W = Wv;  bias = bv;  out = ov;  break;
    default: W = Wsk; bias = bsk; out = osk; break;
  }

  // --- async-stage x tile: 2048 floats, 32 B per thread (2x b128) ---
  {
    unsigned lbase = (unsigned)(size_t)(void*)xs;
    const char* g = (const char*)(x + (size_t)row0 * DIM);
    unsigned off = (unsigned)threadIdx.x * 32u;
    unsigned l0 = lbase + off;
    unsigned l1 = lbase + off + 16u;
    const char* g0 = g + off;
    const char* g1 = g + off + 16u;
    asm volatile("global_load_async_to_lds_b128 %0, %1, off"
                 :: "v"(l0), "v"(g0) : "memory");
    asm volatile("global_load_async_to_lds_b128 %0, %1, off"
                 :: "v"(l1), "v"(g1) : "memory");
  }

  // --- stage W as f16, pre-swizzled into B-fragment order:
  //     slot[((kk*8 + tcol)*32 + fragLane)*16 + i] holds W[kk*32 + fl>>4*16+i][...]
#pragma unroll 4
  for (int t = 0; t < 64; ++t) {
    int idx  = t * 256 + threadIdx.x;       // coalesced over W
    int krow = idx >> 7;                    // 0..127
    int n    = idx & 127;
    int kk   = krow >> 5;
    int w32  = krow & 31;
    int fl   = ((w32 >> 4) << 4) | (n & 15);  // fragment lane
    int i    = w32 & 15;                      // element within v16h
    wls[((((kk << 3) + (n >> 4)) << 5) | fl) * 16 + i] = (_Float16)W[idx];
  }

  asm volatile("s_wait_asynccnt 0x0" ::: "memory");
  __syncthreads();

  v8f acc = {};
#pragma unroll
  for (int kk = 0; kk < 4; ++kk) {
    v16h a = load_a_lds(&xs[kk * 32], DIM);
    v16h b = *(const v16h*)&wls[((((kk << 3) + wave) << 5) | lane) * 16];
    acc = __builtin_amdgcn_wmma_f32_16x16x32_f16(false, a, false, b,
                                                 (short)0, acc, false, false);
  }
  const float bcol = bias[(wave << 4) + (lane & 15)];
#pragma unroll
  for (int r = 0; r < 8; ++r) acc[r] += bcol;
  store_c_f32(out + (size_t)row0 * DIM + (wave << 4), DIM, acc);
}

// ---------------------------------------------------------------------------
// K2: e = edge_attr @ We  (K=32 -> single WMMA per tile), fp16 out.
// Edge tile (16x32 f32) staged via the Tensor Data Mover; We pre-swizzled
// to f16 B-fragment order in LDS.
// ---------------------------------------------------------------------------
__global__ __launch_bounds__(256) void k2_edge_gemm(
    const float* __restrict__ ea, const float* __restrict__ We,
    _Float16* __restrict__ e_out) {
  __shared__ __align__(16) float    es[16 * DEDGE];       // 2 KB
  __shared__ __align__(32) _Float16 wls[8 * 32 * 16];     // 8 KB, frag order

  const int wave = threadIdx.x >> 5;
  const int lane = threadIdx.x & 31;
  const int row0 = blockIdx.x << 4;   // 16 edges

  if (wave == 0) {
    // Build Tensor DMA descriptor D# (groups 0,1) per ISA 08_async_tensor §8.
    unsigned lds_a = (unsigned)(size_t)(void*)es;
    unsigned long long ga = (unsigned long long)(const void*)(ea + (size_t)row0 * DEDGE);
    v4u g0;
    g0[0] = 1u;                                              // count=1, user mode
    g0[1] = lds_a;                                           // lds_addr
    g0[2] = (unsigned)ga;                                    // global_addr[31:0]
    g0[3] = ((unsigned)(ga >> 32) & 0x01FFFFFFu) | (2u << 30); // addr[56:32] | type=2
    v8u g1;
    g1[0] = 0x00020000u;                                     // data_size = 4 bytes
    g1[1] = (unsigned)DEDGE << 16;                           // tensor_dim0 = 32
    g1[2] = ((unsigned)(N_EDGES & 0xFFFF)) << 16;            // tensor_dim1 lo16
    g1[3] = ((unsigned)DEDGE << 16) | (unsigned)(N_EDGES >> 16); // tile_dim0 | dim1 hi
    g1[4] = 16u;                                             // tile_dim1 = 16 rows
    g1[5] = (unsigned)DEDGE;                                 // tensor_dim0_stride
    g1[6] = 0u;
    g1[7] = 0u;
    asm volatile("tensor_load_to_lds %0, %1" :: "s"(g0), "s"(g1) : "memory");
    __builtin_amdgcn_s_wait_tensorcnt(0);
  }

  // stage We (32x128) -> f16 fragment order
#pragma unroll
  for (int t = 0; t < 16; ++t) {
    int idx  = t * 256 + threadIdx.x;
    int krow = idx >> 7;                    // 0..31
    int n    = idx & 127;
    int fl   = ((krow >> 4) << 4) | (n & 15);
    wls[(((n >> 4) << 5) | fl) * 16 + (krow & 15)] = (_Float16)We[idx];
  }
  __syncthreads();

  v16h a = load_a_lds(&es[0], DEDGE);
  v16h b = *(const v16h*)&wls[((wave << 5) | lane) * 16];
  v8f acc = {};
  acc = __builtin_amdgcn_wmma_f32_16x16x32_f16(false, a, false, b,
                                               (short)0, acc, false, false);
  store_c_f16(e_out + (size_t)row0 * DIM + (wave << 4), DIM, acc);
}

// ---------------------------------------------------------------------------
// K3: per (edge, head) attention logit + segment max (ordered-uint atomicMax)
// ---------------------------------------------------------------------------
__global__ __launch_bounds__(256) void k3_logits_max(
    const int* __restrict__ ei, const float* __restrict__ q,
    const float* __restrict__ k, const _Float16* __restrict__ e,
    float* __restrict__ logits, unsigned* __restrict__ mkey) {
  int idx = blockIdx.x * 256 + threadIdx.x;
  if (idx >= N_EDGES * HEADS) return;
  const int eid = idx >> 3, h = idx & 7;
  const int src = ei[eid], tgt = ei[N_EDGES + eid];
  const float*    qp = q + (size_t)tgt * DIM + h * HDIM;
  const float*    kp = k + (size_t)src * DIM + h * HDIM;
  const _Float16* ep = e + (size_t)eid * DIM + h * HDIM;
  float acc = 0.0f;
#pragma unroll
  for (int d = 0; d < HDIM; ++d) acc += qp[d] * (kp[d] + (float)ep[d]);
  const float lg = acc * 0.25f;  // 1/sqrt(D)
  logits[idx] = lg;
  atomicMax(mkey + (size_t)tgt * HEADS + h, enc_f32(lg));
}

// ---------------------------------------------------------------------------
// K4: ex = exp(logit - m[tgt]); segment sum via atomicAdd
// ---------------------------------------------------------------------------
__global__ __launch_bounds__(256) void k4_exp_sum(
    const int* __restrict__ ei, const unsigned* __restrict__ mkey,
    float* __restrict__ logits, float* __restrict__ ssum) {
  int idx = blockIdx.x * 256 + threadIdx.x;
  if (idx >= N_EDGES * HEADS) return;
  const int eid = idx >> 3, h = idx & 7;
  const int tgt = ei[N_EDGES + eid];
  const float m  = dec_f32(mkey[(size_t)tgt * HEADS + h]);
  const float ex = __expf(logits[idx] - m);
  logits[idx] = ex;
  atomicAdd(ssum + (size_t)tgt * HEADS + h, ex);
}

// ---------------------------------------------------------------------------
// K5: agg[tgt] += alpha * (v[src] + e)   (scatter f32 atomics)
// ---------------------------------------------------------------------------
__global__ __launch_bounds__(256) void k5_aggregate(
    const int* __restrict__ ei, const float* __restrict__ v,
    const _Float16* __restrict__ e, const float* __restrict__ logits,
    const float* __restrict__ ssum, float* __restrict__ agg) {
  int idx = blockIdx.x * 256 + threadIdx.x;
  if (idx >= N_EDGES * HEADS) return;
  const int eid = idx >> 3, h = idx & 7;
  const int src = ei[eid], tgt = ei[N_EDGES + eid];
  const float alpha = logits[idx] / ssum[(size_t)tgt * HEADS + h];
  const float*    vp = v   + (size_t)src * DIM + h * HDIM;
  const _Float16* ep = e   + (size_t)eid * DIM + h * HDIM;
  float*          ap = agg + (size_t)tgt * DIM + h * HDIM;
#pragma unroll
  for (int d = 0; d < HDIM; ++d)
    atomicAdd(ap + d, alpha * (vp[d] + (float)ep[d]));
}

// ---------------------------------------------------------------------------
// K6: per-node epilogue: beta gate, conditional LayerNorm, ReLU.
// One wave32 per node; lane owns f = lane + 32*j, j=0..3.
// ---------------------------------------------------------------------------
__global__ __launch_bounds__(256) void k6_finalize(
    const float* __restrict__ agg, const float* __restrict__ skip,
    const float* __restrict__ cond, const float* __restrict__ Wbeta,
    const float* __restrict__ Ws, const float* __restrict__ bs,
    const float* __restrict__ Wb, const float* __restrict__ bb,
    float* __restrict__ out) {
  const int wave = threadIdx.x >> 5;
  const int lane = threadIdx.x & 31;
  const int n = blockIdx.x * 8 + wave;

  float o[4], sk[4];
#pragma unroll
  for (int j = 0; j < 4; ++j) {
    int f = lane + 32 * j;
    o[j]  = agg[(size_t)n * DIM + f];
    sk[j] = skip[(size_t)n * DIM + f];
  }
  // beta gate: sigmoid(<[out, skip, out-skip], Wbeta>)
  float t = 0.0f;
#pragma unroll
  for (int j = 0; j < 4; ++j) {
    int f = lane + 32 * j;
    t += o[j] * Wbeta[f] + sk[j] * Wbeta[DIM + f] + (o[j] - sk[j]) * Wbeta[2 * DIM + f];
  }
  t = wave_sum(t);
  const float g = 1.0f / (1.0f + __expf(-t));
#pragma unroll
  for (int j = 0; j < 4; ++j) o[j] = g * sk[j] + (1.0f - g) * o[j];

  // LayerNorm stats over F=128
  const float mu = wave_sum(o[0] + o[1] + o[2] + o[3]) * (1.0f / 128.0f);
  float var = 0.0f;
#pragma unroll
  for (int j = 0; j < 4; ++j) { float d = o[j] - mu; var += d * d; }
  var = wave_sum(var) * (1.0f / 128.0f);
  const float rstd = rsqrtf(var + 1e-5f);

  // conditional scale/bias (K=16 matvec per lane-column) + ReLU
#pragma unroll
  for (int j = 0; j < 4; ++j) {
    int f = lane + 32 * j;
    float sc = bs[f], bi = bb[f];
#pragma unroll
    for (int c = 0; c < DCOND; ++c) {
      float cv = cond[(size_t)n * DCOND + c];
      sc += cv * Ws[(size_t)c * DIM + f];
      bi += cv * Wb[(size_t)c * DIM + f];
    }
    float r = sc * ((o[j] - mu) * rstd) + bi;
    out[(size_t)n * DIM + f] = fmaxf(r, 0.0f);
  }
}

// ---------------------------------------------------------------------------
extern "C" void kernel_launch(void* const* d_in, const int* in_sizes, int n_in,
                              void* d_out, int out_size, void* d_ws, size_t ws_size,
                              hipStream_t stream) {
  const float* x    = (const float*)d_in[0];
  const int*   ei   = (const int*)d_in[1];     // [2, E] (int32 under default jax x32)
  const float* ea   = (const float*)d_in[2];
  const float* cond = (const float*)d_in[3];
  const float* Wq   = (const float*)d_in[4];   const float* bq  = (const float*)d_in[5];
  const float* Wk   = (const float*)d_in[6];   const float* bk  = (const float*)d_in[7];
  const float* Wv   = (const float*)d_in[8];   const float* bv  = (const float*)d_in[9];
  const float* We   = (const float*)d_in[10];
  const float* Wsk  = (const float*)d_in[11];  const float* bsk = (const float*)d_in[12];
  const float* Wbt  = (const float*)d_in[13];
  const float* Ws   = (const float*)d_in[14];  const float* bs  = (const float*)d_in[15];
  const float* Wb   = (const float*)d_in[16];  const float* bb  = (const float*)d_in[17];
  float* out = (float*)d_out;

  // workspace carve-up (~362 MB)
  float*     q      = (float*)d_ws;
  float*     k      = q    + (size_t)N_NODES * DIM;
  float*     v      = k    + (size_t)N_NODES * DIM;
  float*     skip   = v    + (size_t)N_NODES * DIM;
  _Float16*  e      = (_Float16*)(skip + (size_t)N_NODES * DIM);
  float*     logits = (float*)(e + (size_t)N_EDGES * DIM);
  unsigned*  mkey   = (unsigned*)(logits + (size_t)N_EDGES * HEADS);
  float*     ssum   = (float*)(mkey + (size_t)N_NODES * HEADS);
  float*     agg    = ssum + (size_t)N_NODES * HEADS;

  const int eh_blocks = (N_EDGES * HEADS + 255) / 256;   // 25000

  k0_init<<<(N_NODES * DIM + 255) / 256, 256, 0, stream>>>(agg, ssum, mkey);
  k1_qkvs_gemm<<<dim3(N_NODES / 16, 4), 256, 0, stream>>>(
      x, Wq, bq, Wk, bk, Wv, bv, Wsk, bsk, q, k, v, skip);
  k2_edge_gemm<<<N_EDGES / 16, 256, 0, stream>>>(ea, We, e);
  k3_logits_max<<<eh_blocks, 256, 0, stream>>>(ei, q, k, e, logits, mkey);
  k4_exp_sum<<<eh_blocks, 256, 0, stream>>>(ei, mkey, logits, ssum);
  k5_aggregate<<<eh_blocks, 256, 0, stream>>>(ei, v, e, logits, ssum, agg);
  k6_finalize<<<N_NODES / 8, 256, 0, stream>>>(
      agg, skip, cond, Wbt, Ws, bs, Wb, bb, out);
}
